// EGNN_C_Block_40656160424000
// MI455X (gfx1250) — matrix operand order repaired and checked by reading, not compiled, not run on hardware
//
#include <hip/hip_runtime.h>
#include <math.h>

typedef __attribute__((ext_vector_type(16))) __bf16 v16bf;
typedef __attribute__((ext_vector_type(8)))  float  v8f;
typedef __attribute__((ext_vector_type(2)))  float  v2f;

#define DEV __device__ __forceinline__

constexpr int NN  = 20000;   // nodes
constexpr int EE  = 320000;  // edges
constexpr int NSF = 128;     // scalar features

// ---------------- workspace layout (bytes) ----------------
constexpr size_t SZ_AGG    = (size_t)NN * 128 * 4;
constexpr size_t OFF_MAGGR = 0;
constexpr size_t OFF_PAGGR = OFF_MAGGR + SZ_AGG;
constexpr size_t OFF_DEG   = OFF_PAGGR + SZ_AGG;
constexpr size_t OFF_FRMN1 = (OFF_DEG + (size_t)NN * 4 + 255) & ~(size_t)255;
constexpr size_t SZ_FRMN1  = (size_t)9 * 8 * 512 * 2;   // KB=9, NB=8
constexpr size_t OFF_FRMN2 = OFF_FRMN1 + SZ_FRMN1;
constexpr size_t SZ_FR48   = (size_t)4 * 8 * 512 * 2;   // KB=4, NB=8
constexpr size_t OFF_FRPN1 = OFF_FRMN2 + SZ_FR48;
constexpr size_t OFF_FRPN2 = OFF_FRPN1 + SZ_FR48;
constexpr size_t SZ_FRPN2  = (size_t)4 * 1 * 512 * 2;   // KB=4, NB=1
constexpr size_t OFF_FRUN1 = OFF_FRPN2 + SZ_FRPN2;
constexpr size_t SZ_FRUN1  = (size_t)8 * 8 * 512 * 2;   // KB=8, NB=8
constexpr size_t OFF_FRUN2 = OFF_FRUN1 + SZ_FRUN1;
constexpr size_t OFF_WV8   = OFF_FRUN2 + SZ_FR48;        // [b][o][c] 8*16*16 f32
constexpr size_t OFF_GL8   = OFF_WV8 + (size_t)2048 * 4;
constexpr size_t OFF_GR8   = OFF_GL8 + (size_t)2048 * 4;
constexpr size_t OFF_GO8   = OFF_GR8 + (size_t)2048 * 4; // [b][o][c32] 8*16*32 f32

// ---------------- helpers ----------------
DEV unsigned short f2bf(float f) {            // f32 -> bf16 round-to-nearest-even
  unsigned int u = __float_as_uint(f);
  u += 0x7FFFu + ((u >> 16) & 1u);
  return (unsigned short)(u >> 16);
}

DEV void lds_sync() { asm volatile("s_wait_dscnt 0" ::: "memory"); }

DEV v8f zero8() { v8f z = {0.f,0.f,0.f,0.f,0.f,0.f,0.f,0.f}; return z; }

DEV v8f wmma_bf(v16bf a, v16bf b, v8f c) {
  return __builtin_amdgcn_wmma_f32_16x16x32_bf16(false, a, false, b, (short)0, c, false, false);
}
DEV v8f wmma_f32k4(v2f a, v2f b, v8f c) {
  return __builtin_amdgcn_wmma_f32_16x16x4_f32(false, a, false, b, (short)0, c, false, false);
}

union AFrag {
  v16bf v;
  unsigned int   u32[8];
  unsigned short u16[16];
  uint4          q[2];
};

// A fragment (16x32 bf16) from LDS row-major bf16 tile.
// lane: m = lane&15, h = lane>>4 ; element e: k = k0 + (e<8 ? 8h+e : 16+8h+e-8)
DEV v16bf a_frag_bf16(const unsigned short* X, int strideU16, int m, int k0, int h) {
  AFrag A;
  const unsigned int* base = (const unsigned int*)(X + (size_t)m * strideU16);
  const int i0 = (k0 + 8 * h) >> 1;
  const int i1 = (k0 + 16 + 8 * h) >> 1;
#pragma unroll
  for (int p = 0; p < 4; ++p) { A.u32[p] = base[i0 + p]; A.u32[4 + p] = base[i1 + p]; }
  return A.v;
}

// A fragment from LDS f32 row-major tile (convert to bf16 on the fly)
DEV v16bf a_frag_f32(const float* X, int stride, int m, int k0, int h) {
  AFrag A;
  const float* b0 = X + (size_t)m * stride + k0 + 8 * h;
#pragma unroll
  for (int p = 0; p < 8; ++p) { A.u16[p] = f2bf(b0[p]); A.u16[8 + p] = f2bf(b0[16 + p]); }
  return A.v;
}

// B fragment: pre-packed fragment-major (32 lanes x 16 bf16 = 1024 B per block)
DEV v16bf b_frag(const unsigned short* Wf, int blk, int lane) {
  AFrag B;
  const uint4* p = (const uint4*)(Wf + (size_t)blk * 512 + lane * 16);
  B.q[0] = p[0]; B.q[1] = p[1];
  return B.v;
}

// Cl(3,0) Cayley: blade order 1,e1,e2,e3,e12,e13,e23,e123
DEV void cayley(int i, int j, int& k, float& sg) {
  const int MASK[8] = {0, 1, 2, 4, 3, 5, 6, 7};   // idx<->mask is self-inverse
  int a = MASK[i], b = MASK[j];
  int t = a >> 1, sbit = 0;
  while (t) { sbit ^= (__popc(t & b) & 1); t >>= 1; }
  k = MASK[a ^ b];
  sg = sbit ? -1.0f : 1.0f;
}

// mv_linear (16ch -> 16ch, per-blade weights) via f32 WMMA 16x16x4.
// AT : LDS  blade-major [b][e16][c16]  (contiguous K pairs -> single b64 load)
// WT : glob blade-major [b][o16][c16]
// OUT: LDS  [e][o][b] ; optionally accumulate attr[r] += val^2
template <bool ACC_ATTR>
DEV void mvlinT(const float* AT, const float* WT, float bias_q, float* OUT,
                float* attr, int h, int q) {
#pragma unroll
  for (int b = 0; b < 8; ++b) {
    v8f acc = zero8();
#pragma unroll
    for (int kb = 0; kb < 4; ++kb) {
      const int c0 = kb * 4 + 2 * h;
      v2f af = *(const v2f*)&AT[b * 256 + q * 16 + c0];
      v2f bf = *(const v2f*)&WT[b * 256 + q * 16 + c0];
      acc = wmma_f32k4(af, bf, acc);
    }
#pragma unroll
    for (int r = 0; r < 8; ++r) {
      const float val = acc[r] + (b == 0 ? bias_q : 0.f);
      OUT[((r + 8 * h) * 16 + q) * 8 + b] = val;
      if (ACC_ATTR) attr[r] += val * val;
    }
  }
}

// ---------------- prep kernels ----------------
__global__ void zero_f32(float* __restrict__ p, long long n) {
  long long t = (long long)blockIdx.x * blockDim.x + threadIdx.x;
  if (t < n) p[t] = 0.0f;
}

// pack W (Kreal x NB*16) into bf16 B-fragments: blk=(kb*NB+nb), lane l, elem e,
// k = kb*32 + (e<8 ? 8*(l>>4)+e : 16+8*(l>>4)+e-8) (matches A layout), n = nb*16 + (l&15)
__global__ void pack_b_frags(const float* __restrict__ W, int Kreal, int KB, int NB,
                             unsigned short* __restrict__ dst) {
  int t = blockIdx.x * blockDim.x + threadIdx.x;
  if (t >= KB * NB * 512) return;
  int e = t & 15;
  int l = (t >> 4) & 31;
  int blk = t >> 9;
  int nb = blk % NB, kb = blk / NB;
  int h = l >> 4, n = l & 15;
  int k = kb * 32 + (e < 8 ? 8 * h + e : 16 + 8 * h + (e - 8));
  int cols = NB * 16;
  float val = (k < Kreal) ? W[(size_t)k * cols + nb * 16 + n] : 0.0f;
  dst[t] = f2bf(val);
}

// expand per-grade (16,C,4) weights to blade-major (8,16,C):  dst[b][o][c]
__global__ void expand_w8T(const float* __restrict__ W, int C, float* __restrict__ dst,
                           int total) {
  int t = blockIdx.x * blockDim.x + threadIdx.x;
  if (t >= total) return;
  int c = t % C;
  int o = (t / C) % 16;
  int b = t / (16 * C);
  const int G[8] = {0, 1, 1, 1, 2, 2, 2, 3};
  dst[t] = W[(o * C + c) * 4 + G[b]];
}

// ---------------- edge kernel: 1 wave / 16 edges ----------------
__global__ __launch_bounds__(32) void edge_kernel(
    const float* __restrict__ s, const float* __restrict__ v,
    const int* __restrict__ eidx,
    const float* __restrict__ wv8T, const float* __restrict__ wv_b,
    const unsigned short* __restrict__ frMN1, const float* __restrict__ mn_b1,
    const unsigned short* __restrict__ frMN2, const float* __restrict__ mn_b2,
    const unsigned short* __restrict__ frPN1, const float* __restrict__ pn_b1,
    const unsigned short* __restrict__ frPN2, const float* __restrict__ pn_b2,
    float* __restrict__ maggr, float* __restrict__ paggr, float* __restrict__ deg) {
  __shared__ int sidx[16], ridx[16];
  __shared__ __align__(16) unsigned short Xt[16 * 288];  // bf16 [e][s_send|s_rec|attr|pad]
  __shared__ __align__(16) float vdiffT[8 * 16 * 16];    // blade-major [b][e][c]
  __shared__ __align__(16) float vij[16 * 128];          // [e][o][b]
  __shared__ __align__(16) float mbuf[16 * 128];         // message m (f32)
  __shared__ __align__(16) unsigned short hb16[16 * 128];// hidden relu acts (bf16)

  const int lane = threadIdx.x;
  const int h = lane >> 4, q = lane & 15;
  const int e0 = blockIdx.x * 16;

  // prefetch next tile's indices (global_prefetch_b8)
  __builtin_prefetch(eidx + e0 + 16 + lane, 0, 1);
  __builtin_prefetch(eidx + EE + e0 + 16 + lane, 0, 1);

  if (lane < 16) { sidx[lane] = eidx[e0 + lane]; ridx[lane] = eidx[EE + e0 + lane]; }
  lds_sync();

  // gather s rows (bf16 into Xt) and v rows (fp32 diff into blade-major vdiffT)
  const int cch = lane >> 1;          // channel handled by this lane
  const int bb0 = (lane & 1) * 4;     // blade base 0 or 4
  for (int r = 0; r < 16; ++r) {
    const int sn = sidx[r], rn = ridx[r];
    float4 a = *(const float4*)(s + (size_t)sn * NSF + lane * 4);
    float4 b = *(const float4*)(s + (size_t)rn * NSF + lane * 4);
    uint2 pa, pb;
    pa.x = (unsigned)f2bf(a.x) | ((unsigned)f2bf(a.y) << 16);
    pa.y = (unsigned)f2bf(a.z) | ((unsigned)f2bf(a.w) << 16);
    pb.x = (unsigned)f2bf(b.x) | ((unsigned)f2bf(b.y) << 16);
    pb.y = (unsigned)f2bf(b.z) | ((unsigned)f2bf(b.w) << 16);
    *(uint2*)&Xt[r * 288 + lane * 4] = pa;
    *(uint2*)&Xt[r * 288 + 128 + lane * 4] = pb;

    float4 vr4 = *(const float4*)(v + (size_t)rn * 128 + lane * 4);
    float4 vs4 = *(const float4*)(v + (size_t)sn * 128 + lane * 4);
    vdiffT[(bb0 + 0) * 256 + r * 16 + cch] = vr4.x - vs4.x;
    vdiffT[(bb0 + 1) * 256 + r * 16 + cch] = vr4.y - vs4.y;
    vdiffT[(bb0 + 2) * 256 + r * 16 + cch] = vr4.z - vs4.z;
    vdiffT[(bb0 + 3) * 256 + r * 16 + cch] = vr4.w - vs4.w;
  }
  lds_sync();

  // v_ij = mv_linear(vdiff) via f32 WMMA per blade ; edge_attr = sum_b v_ij^2
  float attr[8];
#pragma unroll
  for (int r = 0; r < 8; ++r) attr[r] = 0.f;
  mvlinT<true>(vdiffT, wv8T, wv_b[q], vij, attr, h, q);
#pragma unroll
  for (int r = 0; r < 8; ++r) Xt[(r + 8 * h) * 288 + 256 + q] = f2bf(attr[r]);
  { uint4 z; z.x = z.y = z.z = z.w = 0u; *(uint4*)&Xt[q * 288 + 272 + 8 * h] = z; }  // pad K
  lds_sync();

  // message net layer 1: (16x288)@(288x128), relu -> hb16
  {
    v16bf A[9];
#pragma unroll
    for (int kb = 0; kb < 9; ++kb) A[kb] = a_frag_bf16(Xt, 288, q, kb * 32, h);
#pragma unroll
    for (int nb = 0; nb < 8; ++nb) {
      v8f acc = zero8();
#pragma unroll
      for (int kb = 0; kb < 9; ++kb) acc = wmma_bf(A[kb], b_frag(frMN1, kb * 8 + nb, lane), acc);
      const float bias = mn_b1[nb * 16 + q];
#pragma unroll
      for (int r = 0; r < 8; ++r) {
        float y = acc[r] + bias;
        hb16[(r + 8 * h) * 128 + nb * 16 + q] = f2bf(y > 0.f ? y : 0.f);
      }
    }
  }
  lds_sync();

  // message net layer 2: (16x128)@(128x128) + b2 -> m (f32)
  {
    v16bf A[4];
#pragma unroll
    for (int kb = 0; kb < 4; ++kb) A[kb] = a_frag_bf16(hb16, 128, q, kb * 32, h);
#pragma unroll
    for (int nb = 0; nb < 8; ++nb) {
      v8f acc = zero8();
#pragma unroll
      for (int kb = 0; kb < 4; ++kb) acc = wmma_bf(A[kb], b_frag(frMN2, kb * 8 + nb, lane), acc);
      const float bias = mn_b2[nb * 16 + q];
#pragma unroll
      for (int r = 0; r < 8; ++r) mbuf[(r + 8 * h) * 128 + nb * 16 + q] = acc[r] + bias;
    }
  }
  lds_sync();

  // pos net layer 1: relu(m @ W1 + b1) -> hb16
  {
    v16bf A[4];
#pragma unroll
    for (int kb = 0; kb < 4; ++kb) A[kb] = a_frag_f32(mbuf, 128, q, kb * 32, h);
#pragma unroll
    for (int nb = 0; nb < 8; ++nb) {
      v8f acc = zero8();
#pragma unroll
      for (int kb = 0; kb < 4; ++kb) acc = wmma_bf(A[kb], b_frag(frPN1, kb * 8 + nb, lane), acc);
      const float bias = pn_b1[nb * 16 + q];
#pragma unroll
      for (int r = 0; r < 8; ++r) {
        float y = acc[r] + bias;
        hb16[(r + 8 * h) * 128 + nb * 16 + q] = f2bf(y > 0.f ? y : 0.f);
      }
    }
  }
  lds_sync();

  // pos net layer 2: (16x128)@(128x16) -> pm (registers)
  float pm[8];
  {
    v16bf A[4];
#pragma unroll
    for (int kb = 0; kb < 4; ++kb) A[kb] = a_frag_bf16(hb16, 128, q, kb * 32, h);
    v8f acc = zero8();
#pragma unroll
    for (int kb = 0; kb < 4; ++kb) acc = wmma_bf(A[kb], b_frag(frPN2, kb, lane), acc);
    const float b2 = pn_b2[q];
#pragma unroll
    for (int r = 0; r < 8; ++r) pm[r] = acc[r] + b2;
  }

  // scatter pos_m = v_ij * pm -> p_aggr ; m -> m_aggr ; deg
#pragma unroll
  for (int r = 0; r < 8; ++r) {
    const int m = r + 8 * h;
    const int rn = ridx[m];
    float* dst = paggr + (size_t)rn * 128 + q * 8;
    const float* src = &vij[(m * 16 + q) * 8];
#pragma unroll
    for (int b = 0; b < 8; ++b) atomicAdd(dst + b, src[b] * pm[r]);
  }
  for (int r = 0; r < 16; ++r) {
    const int rn = ridx[r];
    float4 mv = *(const float4*)&mbuf[r * 128 + lane * 4];
    float* dst = maggr + (size_t)rn * 128 + lane * 4;
    atomicAdd(dst + 0, mv.x); atomicAdd(dst + 1, mv.y);
    atomicAdd(dst + 2, mv.z); atomicAdd(dst + 3, mv.w);
  }
  if (h == 0) atomicAdd(deg + sidx[q], 1.0f);
}

// ---------------- node kernel: 1 wave / 16 nodes ----------------
__global__ __launch_bounds__(32) void node_kernel(
    const float* __restrict__ s, const float* __restrict__ v,
    const unsigned short* __restrict__ frUN1, const float* __restrict__ un_b1,
    const unsigned short* __restrict__ frUN2, const float* __restrict__ un_b2,
    const float* __restrict__ gl8T, const float* __restrict__ gl_b,
    const float* __restrict__ gr8T, const float* __restrict__ gr_b,
    const float* __restrict__ go8T, const float* __restrict__ go_b,
    const float* __restrict__ ln_a,
    const float* __restrict__ maggr, const float* __restrict__ paggr,
    const float* __restrict__ deg,
    float* __restrict__ out_s, float* __restrict__ out_v) {
  __shared__ float invs[16];
  __shared__ __align__(16) unsigned short Xt[16 * 256];   // [node][s | m_aggr*inv] bf16
  __shared__ __align__(16) unsigned short hb16[16 * 128]; // hidden relu acts (bf16)
  __shared__ __align__(16) float pbufT[8 * 16 * 16];      // blade-major [b][e][c]
  __shared__ __align__(16) float vlbuf[16 * 128];         // vl [e][o][b]; reused for vo
  __shared__ __align__(16) float vrbuf[16 * 128];         // vr [e][o][b]
  __shared__ __align__(16) float catbufT[8 * 16 * 32];    // blade-major [b][e][c32]
  __shared__ float nrm[16 * 16];

  const int lane = threadIdx.x;
  const int h = lane >> 4, q = lane & 15;
  const int n0 = blockIdx.x * 16;

  if (lane < 16) invs[lane] = 1.0f / sqrtf(deg[n0 + lane]);
  lds_sync();

  const int cch = lane >> 1;
  const int bb0 = (lane & 1) * 4;
  for (int r = 0; r < 16; ++r) {
    const int nd = n0 + r;
    const float iv = invs[r];
    float4 a = *(const float4*)(s + (size_t)nd * 128 + lane * 4);
    float4 b = *(const float4*)(maggr + (size_t)nd * 128 + lane * 4);
    b.x *= iv; b.y *= iv; b.z *= iv; b.w *= iv;
    uint2 pa, pb;
    pa.x = (unsigned)f2bf(a.x) | ((unsigned)f2bf(a.y) << 16);
    pa.y = (unsigned)f2bf(a.z) | ((unsigned)f2bf(a.w) << 16);
    pb.x = (unsigned)f2bf(b.x) | ((unsigned)f2bf(b.y) << 16);
    pb.y = (unsigned)f2bf(b.z) | ((unsigned)f2bf(b.w) << 16);
    *(uint2*)&Xt[r * 256 + lane * 4] = pa;
    *(uint2*)&Xt[r * 256 + 128 + lane * 4] = pb;

    float4 p4 = *(const float4*)(paggr + (size_t)nd * 128 + lane * 4);
    pbufT[(bb0 + 0) * 256 + r * 16 + cch] = p4.x * iv;
    pbufT[(bb0 + 1) * 256 + r * 16 + cch] = p4.y * iv;
    pbufT[(bb0 + 2) * 256 + r * 16 + cch] = p4.z * iv;
    pbufT[(bb0 + 3) * 256 + r * 16 + cch] = p4.w * iv;
  }
  lds_sync();

  // update net layer 1: (16x256)@(256x128) relu -> hb16
  {
    v16bf A[8];
#pragma unroll
    for (int kb = 0; kb < 8; ++kb) A[kb] = a_frag_bf16(Xt, 256, q, kb * 32, h);
#pragma unroll
    for (int nb = 0; nb < 8; ++nb) {
      v8f acc = zero8();
#pragma unroll
      for (int kb = 0; kb < 8; ++kb) acc = wmma_bf(A[kb], b_frag(frUN1, kb * 8 + nb, lane), acc);
      const float bias = un_b1[nb * 16 + q];
#pragma unroll
      for (int r = 0; r < 8; ++r) {
        float y = acc[r] + bias;
        hb16[(r + 8 * h) * 128 + nb * 16 + q] = f2bf(y > 0.f ? y : 0.f);
      }
    }
  }
  lds_sync();

  // update net layer 2 + residual -> s_new
  {
    v16bf A[4];
#pragma unroll
    for (int kb = 0; kb < 4; ++kb) A[kb] = a_frag_bf16(hb16, 128, q, kb * 32, h);
#pragma unroll
    for (int nb = 0; nb < 8; ++nb) {
      v8f acc = zero8();
#pragma unroll
      for (int kb = 0; kb < 4; ++kb) acc = wmma_bf(A[kb], b_frag(frUN2, kb * 8 + nb, lane), acc);
      const int n = nb * 16 + q;
      const float b2 = un_b2[n];
#pragma unroll
      for (int r = 0; r < 8; ++r) {
        const int nd = n0 + r + 8 * h;
        out_s[(size_t)nd * 128 + n] = s[(size_t)nd * 128 + n] + acc[r] + b2;
      }
    }
  }

  // vl / vr = mv_linear(p) via f32 WMMA (blade-major operands)
  mvlinT<false>(pbufT, gl8T, gl_b[q], vlbuf, nullptr, h, q);
  mvlinT<false>(pbufT, gr8T, gr_b[q], vrbuf, nullptr, h, q);
  lds_sync();

  // geometric product gp(vl,vr) -> catT[ch 0..15], copy p -> catT[ch 16..31]
#pragma unroll
  for (int t = 0; t < 8; ++t) {
    const int oc = 8 * h + t;
    float av[8], bv[8], oz[8];
    {
      float4 a0 = *(const float4*)&vlbuf[(q * 16 + oc) * 8];
      float4 a1 = *(const float4*)&vlbuf[(q * 16 + oc) * 8 + 4];
      float4 b0 = *(const float4*)&vrbuf[(q * 16 + oc) * 8];
      float4 b1 = *(const float4*)&vrbuf[(q * 16 + oc) * 8 + 4];
      av[0]=a0.x; av[1]=a0.y; av[2]=a0.z; av[3]=a0.w;
      av[4]=a1.x; av[5]=a1.y; av[6]=a1.z; av[7]=a1.w;
      bv[0]=b0.x; bv[1]=b0.y; bv[2]=b0.z; bv[3]=b0.w;
      bv[4]=b1.x; bv[5]=b1.y; bv[6]=b1.z; bv[7]=b1.w;
    }
#pragma unroll
    for (int i = 0; i < 8; ++i) oz[i] = 0.f;
#pragma unroll
    for (int i = 0; i < 8; ++i)
#pragma unroll
      for (int j = 0; j < 8; ++j) {
        int k; float sg;
        cayley(i, j, k, sg);
        oz[k] += sg * av[i] * bv[j];
      }
#pragma unroll
    for (int bb = 0; bb < 8; ++bb) {
      catbufT[bb * 512 + q * 32 + oc] = oz[bb];
      catbufT[bb * 512 + q * 32 + 16 + oc] = pbufT[bb * 256 + q * 16 + oc];
    }
  }
  lds_sync();

  // vo = mv_linear(cat, go) : K=32 per blade via f32 WMMA ; D -> vlbuf (reuse, [e][o][b])
  float ssq[8];
#pragma unroll
  for (int r = 0; r < 8; ++r) ssq[r] = 0.f;
  const float gob = go_b[q];
#pragma unroll
  for (int b = 0; b < 8; ++b) {
    v8f acc = zero8();
#pragma unroll
    for (int kb = 0; kb < 8; ++kb) {
      const int c0 = kb * 4 + 2 * h;
      v2f af = *(const v2f*)&catbufT[b * 512 + q * 32 + c0];
      v2f bf = *(const v2f*)&go8T[b * 512 + q * 32 + c0];
      acc = wmma_f32k4(af, bf, acc);
    }
#pragma unroll
    for (int r = 0; r < 8; ++r) {
      const float val = acc[r] + (b == 0 ? gob : 0.f);
      vlbuf[((r + 8 * h) * 16 + q) * 8 + b] = val;
      ssq[r] += val * val;
    }
  }
#pragma unroll
  for (int r = 0; r < 8; ++r) nrm[(r + 8 * h) * 16 + q] = sqrtf(ssq[r]);
  lds_sync();

  // MVLayerNorm + residual -> v_new
  {
    float nsum = 0.f;
#pragma unroll
    for (int o = 0; o < 16; ++o) nsum += nrm[q * 16 + o];
    const float nval = nsum * (1.0f / 16.0f) + 1e-6f;
    const int nd = n0 + q;
#pragma unroll
    for (int t = 0; t < 8; ++t) {
      const int oc = 8 * h + t;
      const float la = ln_a[oc];
#pragma unroll
      for (int bb = 0; bb < 8; ++bb) {
        const float vo = vlbuf[(q * 16 + oc) * 8 + bb];
        out_v[(size_t)nd * 128 + oc * 8 + bb] = la * vo / nval + v[(size_t)nd * 128 + oc * 8 + bb];
      }
    }
  }
}

// ---------------- launch ----------------
extern "C" void kernel_launch(void* const* d_in, const int* in_sizes, int n_in,
                              void* d_out, int out_size, void* d_ws, size_t ws_size,
                              hipStream_t stream) {
  (void)in_sizes; (void)n_in; (void)out_size; (void)ws_size;
  const float* s     = (const float*)d_in[0];
  const float* v     = (const float*)d_in[1];
  const int*   eidx  = (const int*)d_in[2];
  const float* Wv_w  = (const float*)d_in[3];
  const float* Wv_b  = (const float*)d_in[4];
  const float* mn_W1 = (const float*)d_in[5];
  const float* mn_b1 = (const float*)d_in[6];
  const float* mn_W2 = (const float*)d_in[7];
  const float* mn_b2 = (const float*)d_in[8];
  const float* pn_W1 = (const float*)d_in[9];
  const float* pn_b1 = (const float*)d_in[10];
  const float* pn_W2 = (const float*)d_in[11];
  const float* pn_b2 = (const float*)d_in[12];
  const float* un_W1 = (const float*)d_in[13];
  const float* un_b1 = (const float*)d_in[14];
  const float* un_W2 = (const float*)d_in[15];
  const float* un_b2 = (const float*)d_in[16];
  const float* gl_w  = (const float*)d_in[17];
  const float* gl_b  = (const float*)d_in[18];
  const float* gr_w  = (const float*)d_in[19];
  const float* gr_b  = (const float*)d_in[20];
  const float* go_w  = (const float*)d_in[21];
  const float* go_b  = (const float*)d_in[22];
  const float* ln_a  = (const float*)d_in[23];

  char* ws = (char*)d_ws;
  float* maggr = (float*)(ws + OFF_MAGGR);
  float* paggr = (float*)(ws + OFF_PAGGR);
  float* deg   = (float*)(ws + OFF_DEG);
  unsigned short* frMN1 = (unsigned short*)(ws + OFF_FRMN1);
  unsigned short* frMN2 = (unsigned short*)(ws + OFF_FRMN2);
  unsigned short* frPN1 = (unsigned short*)(ws + OFF_FRPN1);
  unsigned short* frPN2 = (unsigned short*)(ws + OFF_FRPN2);
  unsigned short* frUN1 = (unsigned short*)(ws + OFF_FRUN1);
  unsigned short* frUN2 = (unsigned short*)(ws + OFF_FRUN2);
  float* wv8T = (float*)(ws + OFF_WV8);
  float* gl8T = (float*)(ws + OFF_GL8);
  float* gr8T = (float*)(ws + OFF_GR8);
  float* go8T = (float*)(ws + OFF_GO8);

  {
    long long zn = (long long)NN * 128 * 2 + NN;  // maggr+paggr+deg contiguous
    zero_f32<<<(unsigned)((zn + 255) / 256), 256, 0, stream>>>(maggr, zn);
  }

  pack_b_frags<<<(9 * 8 * 512 + 255) / 256, 256, 0, stream>>>(mn_W1, 272, 9, 8, frMN1);
  pack_b_frags<<<(4 * 8 * 512 + 255) / 256, 256, 0, stream>>>(mn_W2, 128, 4, 8, frMN2);
  pack_b_frags<<<(4 * 8 * 512 + 255) / 256, 256, 0, stream>>>(pn_W1, 128, 4, 8, frPN1);
  pack_b_frags<<<(4 * 1 * 512 + 255) / 256, 256, 0, stream>>>(pn_W2, 128, 4, 1, frPN2);
  pack_b_frags<<<(8 * 8 * 512 + 255) / 256, 256, 0, stream>>>(un_W1, 256, 8, 8, frUN1);
  pack_b_frags<<<(4 * 8 * 512 + 255) / 256, 256, 0, stream>>>(un_W2, 128, 4, 8, frUN2);

  expand_w8T<<<8, 256, 0, stream>>>(Wv_w, 16, wv8T, 2048);
  expand_w8T<<<8, 256, 0, stream>>>(gl_w, 16, gl8T, 2048);
  expand_w8T<<<8, 256, 0, stream>>>(gr_w, 16, gr8T, 2048);
  expand_w8T<<<16, 256, 0, stream>>>(go_w, 32, go8T, 4096);

  edge_kernel<<<EE / 16, 32, 0, stream>>>(s, v, eidx, wv8T, Wv_b,
      frMN1, mn_b1, frMN2, mn_b2, frPN1, pn_b1, frPN2, pn_b2,
      maggr, paggr, deg);

  float* out_s = (float*)d_out;
  float* out_v = out_s + (size_t)NN * 128;
  node_kernel<<<NN / 16, 32, 0, stream>>>(s, v, frUN1, un_b1, frUN2, un_b2,
      gl8T, gl_b, gr8T, gr_b, go8T, go_b, ln_a, maggr, paggr, deg, out_s, out_v);
}
